// GPT_66846870995397
// MI455X (gfx1250) — matrix operand order repaired and checked by reading, not compiled, not run on hardware
//
#include <hip/hip_runtime.h>
#include <hip/hip_bf16.h>

// ---------------- problem constants ----------------
constexpr int N_  = 4;
constexpr int S_  = 1024;
constexpr int E_  = 1024;
constexpr int H_  = 16;
constexpr int HD_ = 64;
constexpr int FF_ = 4096;
constexpr int NS_ = N_ * S_;   // 4096 rows of the token stream

typedef __bf16 bf16_t;
typedef __attribute__((ext_vector_type(8)))  __bf16 v8bf;
typedef __attribute__((ext_vector_type(16))) __bf16 v16bf;
typedef __attribute__((ext_vector_type(8)))  float  v8f;

// ---------------- WMMA fragment helpers ----------------
// A fragment (16x32, bf16, row-major source, lda in elements).
// Layout used here (contiguous k, not the ISA table order): lane = half*16+idx holds row (idx), elements 0..15 = K[kBase+half*16 ..]
// (16 consecutive K)  -> two contiguous 16B loads.
__device__ __forceinline__ v16bf load_a_frag(const bf16_t* __restrict__ base,
                                             int lda, int row0, int kBase) {
    const int lane = threadIdx.x & 31;
    const int half = lane >> 4, idx = lane & 15;
    const bf16_t* p = base + (size_t)(row0 + idx) * lda + kBase + half * 16;
    v8bf lo = *reinterpret_cast<const v8bf*>(p);
    v8bf hi = *reinterpret_cast<const v8bf*>(p + 8);
    return __builtin_shufflevector(lo, hi, 0,1,2,3,4,5,6,7,8,9,10,11,12,13,14,15);
}

// B fragment (32x16, bf16) from Bt stored as [Ndim][K] row-major (i.e. B transposed).
// Layout used here (contiguous k, same as A): lane holds column (idx), 16 consecutive K values starting at kBase+half*16.
__device__ __forceinline__ v16bf load_b_frag(const bf16_t* __restrict__ bt,
                                             int ldbt, int col0, int kBase) {
    const int lane = threadIdx.x & 31;
    const int half = lane >> 4, idx = lane & 15;
    const bf16_t* p = bt + (size_t)(col0 + idx) * ldbt + kBase + half * 16;
    return *reinterpret_cast<const v16bf*>(p);
}

__device__ __forceinline__ v8f wmma_bf16(v16bf a, v16bf b, v8f c) {
    return __builtin_amdgcn_wmma_f32_16x16x32_bf16(false, a, false, b,
                                                   (short)0, c, false, false);
}

// gfx1250 async global->LDS copy (16B per lane), tracked by ASYNCcnt.
// lds_addr is the byte offset within the wave's LDS allocation
// (= low 32 bits of the generic shared-memory address).
__device__ __forceinline__ void async_copy_b128(unsigned lds_addr,
                                                const bf16_t* gsrc) {
    asm volatile("global_load_async_to_lds_b128 %0, %1, off"
                 :: "v"(lds_addr), "v"(gsrc)
                 : "memory");
}

// ---------------- generic bf16 WMMA GEMM with async-LDS double buffering ----
// C[M][Ndim] = act( A[M][K] @ Bt[Ndim][K]^T + bias + residual )
// Block = 128 threads = 4 waves (2x2), each wave computes a 64x64 tile,
// block tile = 128x128. Per K-step of 32, a 128x32 A tile and a 128x32 Bt tile
// are staged in LDS via global_load_async_to_lds_b128 (double buffered,
// overlapping the 16 WMMAs of the previous step); s_wait_asynccnt + barrier
// close the pipeline stage. grid = (M/128, Ndim/128); all dims tile-aligned.
__global__ __launch_bounds__(128) void gemm_bf16_kernel(
    const bf16_t* __restrict__ A, int lda,
    const bf16_t* __restrict__ Bt, int ldbt,
    const float* __restrict__ bias,
    const float* __restrict__ residual,
    float* __restrict__ outF, bf16_t* __restrict__ outB,
    int Ndim, int K, int act)
{
    constexpr int TILE_ELEMS = 128 * 32;          // one staged matrix tile
    constexpr unsigned MATB  = TILE_ELEMS * 2;    // bytes per tile (8KB)
    __shared__ __align__(16) bf16_t smem[2 * 2 * TILE_ELEMS];  // [buf][A|B][128][32]

    const int w    = threadIdx.x >> 5;
    const int lane = threadIdx.x & 31;
    const int half = lane >> 4, idx = lane & 15;
    const int wr   = (w >> 1) * 64;               // wave row offset in block tile
    const int wc   = (w & 1) * 64;                // wave col offset in block tile
    const int wr0  = blockIdx.x * 128 + wr;
    const int wc0  = blockIdx.y * 128 + wc;

    const bf16_t* Atile = A  + (size_t)blockIdx.x * 128 * lda;
    const bf16_t* Btile = Bt + (size_t)blockIdx.y * 128 * ldbt;
    const unsigned sbase = (unsigned)(size_t)(const void*)smem;
    const int t = threadIdx.x;                    // 0..127 -> staged tile row

    // issue one buffer fill: 4 x 16B segments of A row t + 4 of Bt row t
    auto fill = [&](int buf, int k) {
        const bf16_t* sa = Atile + (size_t)t * lda + k;
        const bf16_t* sb = Btile + (size_t)t * ldbt + k;
        const unsigned da = sbase + (unsigned)buf * 2u * MATB + (unsigned)t * 64u;
        const unsigned db = da + MATB;
        #pragma unroll
        for (int seg = 0; seg < 4; ++seg) async_copy_b128(da + seg * 16, sa + seg * 8);
        #pragma unroll
        for (int seg = 0; seg < 4; ++seg) async_copy_b128(db + seg * 16, sb + seg * 8);
    };

    fill(0, 0);

    v8f acc[4][4] = {};
    int buf = 0;
    for (int k = 0; k < K; k += 32) {
        if (k + 32 < K) {
            fill(buf ^ 1, k + 32);                      // 8 new async ops in flight
            asm volatile("s_wait_asynccnt 0x8" ::: "memory");  // prev 8 complete
        } else {
            asm volatile("s_wait_asynccnt 0x0" ::: "memory");
        }
        __syncthreads();                                // tile visible to all waves

        const bf16_t* Abuf = smem + buf * (2 * TILE_ELEMS);
        const bf16_t* Bbuf = Abuf + TILE_ELEMS;
        v16bf a[4], b[4];
        #pragma unroll
        for (int mi = 0; mi < 4; ++mi) a[mi] = load_a_frag(Abuf, 32, wr + 16 * mi, 0);
        #pragma unroll
        for (int ni = 0; ni < 4; ++ni) b[ni] = load_b_frag(Bbuf, 32, wc + 16 * ni, 0);
        #pragma unroll
        for (int mi = 0; mi < 4; ++mi)
            #pragma unroll
            for (int ni = 0; ni < 4; ++ni)
                acc[mi][ni] = wmma_bf16(a[mi], b[ni], acc[mi][ni]);

        __syncthreads();                                // reads done before overwrite
        buf ^= 1;
    }

    #pragma unroll
    for (int mi = 0; mi < 4; ++mi) {
        #pragma unroll
        for (int ni = 0; ni < 4; ++ni) {
            const int c  = wc0 + ni * 16 + idx;
            const float bv = bias ? bias[c] : 0.0f;
            #pragma unroll
            for (int j = 0; j < 8; ++j) {
                const int r = wr0 + mi * 16 + half * 8 + j;  // C layout: M = 8*half + j
                float v = acc[mi][ni][j] + bv;
                if (residual) v += residual[(size_t)r * Ndim + c];
                if (act) v = v > 0.0f ? v : 0.05f * v;       // leaky relu 0.05
                if (outF) outF[(size_t)r * Ndim + c] = v;
                if (outB) outB[(size_t)r * Ndim + c] = (bf16_t)v;
            }
        }
    }
}

// ---------------- per-head QKV projection ----------------
// For (n,h): out = x[n, :, h*64:(h+1)*64] @ W (64x64, shared across heads).
// z=0 -> Q (s,d), z=1 -> K (s,d), z=2 -> V stored transposed (d,s) for the PV GEMM.
__global__ __launch_bounds__(128) void qkv_kernel(
    const bf16_t* __restrict__ xb,
    const bf16_t* __restrict__ Wqt, const bf16_t* __restrict__ Wkt,
    const bf16_t* __restrict__ Wvt,
    bf16_t* __restrict__ Qo, bf16_t* __restrict__ Ko, bf16_t* __restrict__ VTo)
{
    const int w    = threadIdx.x >> 5;
    const int lane = threadIdx.x & 31;
    const int half = lane >> 4, idx = lane & 15;
    const int nh = blockIdx.y;
    const int n  = nh >> 4, h = nh & 15;
    const int r0 = blockIdx.x * 256 + w * 64;
    const int which = blockIdx.z;
    const bf16_t* Wt    = which == 0 ? Wqt : which == 1 ? Wkt : Wvt;
    const bf16_t* Abase = xb + (size_t)n * S_ * E_ + h * HD_;

    v8f acc[4][4] = {};
    #pragma unroll
    for (int k = 0; k < HD_; k += 32) {
        v16bf a[4], b[4];
        #pragma unroll
        for (int mi = 0; mi < 4; ++mi) a[mi] = load_a_frag(Abase, E_, r0 + 16 * mi, k);
        #pragma unroll
        for (int ni = 0; ni < 4; ++ni) b[ni] = load_b_frag(Wt, HD_, 16 * ni, k);
        #pragma unroll
        for (int mi = 0; mi < 4; ++mi)
            #pragma unroll
            for (int ni = 0; ni < 4; ++ni)
                acc[mi][ni] = wmma_bf16(a[mi], b[ni], acc[mi][ni]);
    }

    if (which < 2) {
        bf16_t* outp = (which == 0 ? Qo : Ko) + (size_t)nh * S_ * HD_;
        #pragma unroll
        for (int mi = 0; mi < 4; ++mi)
            #pragma unroll
            for (int ni = 0; ni < 4; ++ni)
                #pragma unroll
                for (int j = 0; j < 8; ++j) {
                    const int r = r0 + mi * 16 + half * 8 + j;
                    const int c = ni * 16 + idx;
                    outp[(size_t)r * HD_ + c] = (bf16_t)acc[mi][ni][j];
                }
    } else {
        bf16_t* outp = VTo + (size_t)nh * HD_ * S_;
        #pragma unroll
        for (int mi = 0; mi < 4; ++mi)
            #pragma unroll
            for (int ni = 0; ni < 4; ++ni)
                #pragma unroll
                for (int j = 0; j < 8; ++j) {
                    const int r = r0 + mi * 16 + half * 8 + j;
                    const int c = ni * 16 + idx;
                    outp[(size_t)c * S_ + r] = (bf16_t)acc[mi][ni][j];
                }
    }
}

// ---------------- flash attention ----------------
// grid = (S/64, N*H), 128 threads = 4 waves; wave owns 16 query rows.
// scores = Q K^T * (1/sqrt(E)); online softmax; O = P V; output concat heads (bf16).
__global__ __launch_bounds__(128) void attn_kernel(
    const bf16_t* __restrict__ Q, const bf16_t* __restrict__ Kk,
    const bf16_t* __restrict__ VT, bf16_t* __restrict__ AO)
{
    __shared__ __align__(16) bf16_t ldsP[4 * 16 * 80];  // per-wave 16x64 P tile, stride 80
    const int w    = threadIdx.x >> 5;
    const int lane = threadIdx.x & 31;
    const int half = lane >> 4, idx = lane & 15;
    const int nh = blockIdx.y;
    const int n  = nh >> 4, h = nh & 15;
    const int q0 = blockIdx.x * 64 + w * 16;
    const bf16_t* Qb = Q  + (size_t)nh * S_ * HD_;
    const bf16_t* Kb = Kk + (size_t)nh * S_ * HD_;
    const bf16_t* Vb = VT + (size_t)nh * HD_ * S_;
    bf16_t* Pl = &ldsP[w * 16 * 80];
    const float scale = 0.03125f;  // 1/sqrt(1024)

    v16bf aq[2];
    aq[0] = load_a_frag(Qb, HD_, q0, 0);
    aq[1] = load_a_frag(Qb, HD_, q0, 32);

    float mrun[8], lrun[8];
    #pragma unroll
    for (int j = 0; j < 8; ++j) { mrun[j] = -1e30f; lrun[j] = 0.0f; }
    v8f o[4] = {};

    for (int kb = 0; kb < S_; kb += 64) {
        v8f sc[4] = {};
        #pragma unroll
        for (int ni = 0; ni < 4; ++ni)
            #pragma unroll
            for (int ks = 0; ks < 2; ++ks) {
                v16bf bk = load_b_frag(Kb, HD_, kb + ni * 16, ks * 32);
                sc[ni] = wmma_bf16(aq[ks], bk, sc[ni]);
            }

        // online softmax over the 64 keys of this block; row = 8*half + j,
        // keys striped over the 16 lanes of each half -> shfl_xor 1,2,4,8.
        float mnew[8], rsc[8], rowsum[8];
        #pragma unroll
        for (int j = 0; j < 8; ++j) {
            float mx = sc[0][j];
            #pragma unroll
            for (int ni = 1; ni < 4; ++ni) mx = fmaxf(mx, sc[ni][j]);
            mx *= scale;
            #pragma unroll
            for (int m = 1; m < 16; m <<= 1) mx = fmaxf(mx, __shfl_xor(mx, m, 32));
            mnew[j]   = fmaxf(mrun[j], mx);
            rsc[j]    = __expf(mrun[j] - mnew[j]);
            mrun[j]   = mnew[j];
            rowsum[j] = 0.0f;
        }
        #pragma unroll
        for (int ni = 0; ni < 4; ++ni)
            #pragma unroll
            for (int j = 0; j < 8; ++j) {
                float p = __expf(sc[ni][j] * scale - mnew[j]);
                sc[ni][j] = p;
                rowsum[j] += p;
            }
        #pragma unroll
        for (int j = 0; j < 8; ++j) {
            float rs = rowsum[j];
            #pragma unroll
            for (int m = 1; m < 16; m <<= 1) rs += __shfl_xor(rs, m, 32);
            lrun[j] = lrun[j] * rsc[j] + rs;
        }
        #pragma unroll
        for (int di = 0; di < 4; ++di)
            #pragma unroll
            for (int j = 0; j < 8; ++j) o[di][j] *= rsc[j];

        // P (C layout) -> LDS bf16 -> re-enter as A fragment for P @ V.
        #pragma unroll
        for (int ni = 0; ni < 4; ++ni)
            #pragma unroll
            for (int j = 0; j < 8; ++j)
                Pl[(half * 8 + j) * 80 + ni * 16 + idx] = (bf16_t)sc[ni][j];
        __syncthreads();
        #pragma unroll
        for (int ks = 0; ks < 2; ++ks) {
            v16bf ap = load_a_frag(Pl, 80, 0, ks * 32);
            #pragma unroll
            for (int di = 0; di < 4; ++di) {
                v16bf bv = load_b_frag(Vb, S_, di * 16, kb + ks * 32);
                o[di] = wmma_bf16(ap, bv, o[di]);
            }
        }
        __syncthreads();
    }

    #pragma unroll
    for (int di = 0; di < 4; ++di)
        #pragma unroll
        for (int j = 0; j < 8; ++j) {
            const float v = o[di][j] / lrun[j];
            const int s = q0 + half * 8 + j;
            const int d = di * 16 + idx;
            AO[((size_t)n * S_ + s) * E_ + h * HD_ + d] = (bf16_t)v;
        }
}

// ---------------- layernorm over E=1024 ----------------
__global__ __launch_bounds__(256) void layernorm_kernel(
    const float* __restrict__ in, const float* __restrict__ g,
    const float* __restrict__ b, float* __restrict__ outF, bf16_t* __restrict__ outB)
{
    __shared__ float red[256];
    const int row = blockIdx.x;
    const float* x = in + (size_t)row * E_;
    float v[4];
    float s = 0.0f;
    #pragma unroll
    for (int i = 0; i < 4; ++i) { v[i] = x[threadIdx.x + i * 256]; s += v[i]; }
    red[threadIdx.x] = s; __syncthreads();
    for (int off = 128; off > 0; off >>= 1) {
        if ((int)threadIdx.x < off) red[threadIdx.x] += red[threadIdx.x + off];
        __syncthreads();
    }
    const float mean = red[0] * (1.0f / E_);
    __syncthreads();
    float sq = 0.0f;
    #pragma unroll
    for (int i = 0; i < 4; ++i) { float d = v[i] - mean; sq += d * d; }
    red[threadIdx.x] = sq; __syncthreads();
    for (int off = 128; off > 0; off >>= 1) {
        if ((int)threadIdx.x < off) red[threadIdx.x] += red[threadIdx.x + off];
        __syncthreads();
    }
    const float rstd = rsqrtf(red[0] * (1.0f / E_) + 1e-5f);
    #pragma unroll
    for (int i = 0; i < 4; ++i) {
        const int c = threadIdx.x + i * 256;
        const float y = (v[i] - mean) * rstd * g[c] + b[c];
        outF[(size_t)row * E_ + c] = y;
        if (outB) outB[(size_t)row * E_ + c] = (bf16_t)y;
    }
}

// ---------------- x + positional encoding ----------------
__global__ __launch_bounds__(256) void addpe_kernel(
    const float* __restrict__ x, float* __restrict__ outF, bf16_t* __restrict__ outB)
{
    const int row = blockIdx.x;     // n*S + s
    const int s = row & (S_ - 1);
    #pragma unroll
    for (int i = 0; i < 4; ++i) {
        const int c = threadIdx.x + i * 256;
        const int p = c >> 1;
        const float freq = __powf(10000.0f, -(float)(2 * p) / (float)E_);
        const float ang  = (float)s * freq;
        const float pe   = (c & 1) ? __cosf(ang) : __sinf(ang);
        const float v = x[(size_t)row * E_ + c] + pe;
        outF[(size_t)row * E_ + c] = v;
        outB[(size_t)row * E_ + c] = (bf16_t)v;
    }
}

// ---------------- weight transpose + bf16 convert: Wt[n][k] = W[k][n] ----------------
__global__ __launch_bounds__(256) void transpose_bf16_kernel(
    const float* __restrict__ W, bf16_t* __restrict__ Wt, int K, int Ndim)
{
    const int i = blockIdx.x * 256 + threadIdx.x;
    if (i >= K * Ndim) return;
    const int k = i / Ndim, n = i - k * Ndim;
    Wt[(size_t)n * K + k] = (bf16_t)W[(size_t)k * Ndim + n];
}

// ---------------- classifier head: logits[n] = out[n,-1,:] . Wc + bc ----------------
__global__ __launch_bounds__(256) void cls_kernel(
    const float* __restrict__ xf, const float* __restrict__ Wc,
    const float* __restrict__ bc, float* __restrict__ out)
{
    __shared__ float red[256];
    const int n = blockIdx.x;
    const float* xr = xf + ((size_t)n * S_ + (S_ - 1)) * E_;
    float s = 0.0f;
    #pragma unroll
    for (int i = 0; i < 4; ++i) {
        const int c = threadIdx.x + i * 256;
        s += xr[c] * Wc[c];
    }
    red[threadIdx.x] = s; __syncthreads();
    for (int off = 128; off > 0; off >>= 1) {
        if ((int)threadIdx.x < off) red[threadIdx.x] += red[threadIdx.x + off];
        __syncthreads();
    }
    if (threadIdx.x == 0) out[n] = red[0] + bc[0];
}

// ---------------- host orchestration ----------------
extern "C" void kernel_launch(void* const* d_in, const int* in_sizes, int n_in,
                              void* d_out, int out_size, void* d_ws, size_t ws_size,
                              hipStream_t stream) {
    const float* x     = (const float*)d_in[0];
    const float* Wq    = (const float*)d_in[1];
    const float* Wk    = (const float*)d_in[2];
    const float* Wv    = (const float*)d_in[3];
    const float* Wo    = (const float*)d_in[4];
    const float* bo    = (const float*)d_in[5];
    const float* ln1g  = (const float*)d_in[6];
    const float* ln1b  = (const float*)d_in[7];
    const float* W1    = (const float*)d_in[8];
    const float* b1    = (const float*)d_in[9];
    const float* W2    = (const float*)d_in[10];
    const float* b2    = (const float*)d_in[11];
    const float* ln2g  = (const float*)d_in[12];
    const float* ln2b  = (const float*)d_in[13];
    const float* Wc    = (const float*)d_in[14];
    const float* bc    = (const float*)d_in[15];

    char* ws = (char*)d_ws;
    size_t off = 0;
    auto take = [&](size_t bytes) -> void* {
        void* p = ws + off;
        off += (bytes + 255) & ~(size_t)255;
        return p;
    };
    float*  xf   = (float*)take((size_t)NS_ * E_ * 4);   // residual stream fp32
    bf16_t* xb   = (bf16_t*)take((size_t)NS_ * E_ * 2);  // residual stream bf16
    bf16_t* qb   = (bf16_t*)take((size_t)NS_ * E_ * 2);
    bf16_t* kb   = (bf16_t*)take((size_t)NS_ * E_ * 2);
    bf16_t* vtb  = (bf16_t*)take((size_t)NS_ * E_ * 2);
    bf16_t* aob  = (bf16_t*)take((size_t)NS_ * E_ * 2);
    float*  tmp  = (float*)take((size_t)NS_ * E_ * 4);
    float*  x1f  = (float*)take((size_t)NS_ * E_ * 4);
    bf16_t* x1b  = (bf16_t*)take((size_t)NS_ * E_ * 2);
    bf16_t* ff1b = (bf16_t*)take((size_t)NS_ * FF_ * 2);
    bf16_t* wqt  = (bf16_t*)take((size_t)HD_ * HD_ * 2);
    bf16_t* wkt  = (bf16_t*)take((size_t)HD_ * HD_ * 2);
    bf16_t* wvt  = (bf16_t*)take((size_t)HD_ * HD_ * 2);
    bf16_t* wot  = (bf16_t*)take((size_t)E_ * E_ * 2);
    bf16_t* w1t  = (bf16_t*)take((size_t)E_ * FF_ * 2);
    bf16_t* w2t  = (bf16_t*)take((size_t)FF_ * E_ * 2);
    (void)ws_size; (void)in_sizes; (void)n_in; (void)out_size;

    addpe_kernel<<<NS_, 256, 0, stream>>>(x, xf, xb);

    for (int l = 0; l < 4; ++l) {
        auto tgrid = [](int elems) { return dim3((elems + 255) / 256); };
        transpose_bf16_kernel<<<tgrid(HD_ * HD_), 256, 0, stream>>>(Wq + (size_t)l * HD_ * HD_, wqt, HD_, HD_);
        transpose_bf16_kernel<<<tgrid(HD_ * HD_), 256, 0, stream>>>(Wk + (size_t)l * HD_ * HD_, wkt, HD_, HD_);
        transpose_bf16_kernel<<<tgrid(HD_ * HD_), 256, 0, stream>>>(Wv + (size_t)l * HD_ * HD_, wvt, HD_, HD_);
        transpose_bf16_kernel<<<tgrid(E_ * E_),  256, 0, stream>>>(Wo + (size_t)l * E_ * E_,  wot, E_, E_);
        transpose_bf16_kernel<<<tgrid(E_ * FF_), 256, 0, stream>>>(W1 + (size_t)l * E_ * FF_, w1t, E_, FF_);
        transpose_bf16_kernel<<<tgrid(FF_ * E_), 256, 0, stream>>>(W2 + (size_t)l * FF_ * E_, w2t, FF_, E_);

        // Q/K/V projections (per-head 64x64 weight shared across heads)
        qkv_kernel<<<dim3(S_ / 256, N_ * H_, 3), 128, 0, stream>>>(
            xb, wqt, wkt, wvt, qb, kb, vtb);

        // flash attention, writes concat-head bf16 output
        attn_kernel<<<dim3(S_ / 64, N_ * H_), 128, 0, stream>>>(qb, kb, vtb, aob);

        // ao @ Wo + bo + residual(out)  -> tmp (fp32)
        gemm_bf16_kernel<<<dim3(NS_ / 128, E_ / 128), 128, 0, stream>>>(
            aob, E_, wot, E_, bo + (size_t)l * E_, xf, tmp, nullptr, E_, E_, 0);

        layernorm_kernel<<<NS_, 256, 0, stream>>>(
            tmp, ln1g + (size_t)l * E_, ln1b + (size_t)l * E_, x1f, x1b);

        // leaky_relu(x1 @ W1 + b1) -> ff1b (bf16 only)
        gemm_bf16_kernel<<<dim3(NS_ / 128, FF_ / 128), 128, 0, stream>>>(
            x1b, E_, w1t, E_, b1 + (size_t)l * FF_, nullptr, nullptr, ff1b, FF_, E_, 1);

        // ff1 @ W2 + b2 + residual(x1) -> tmp (fp32)
        gemm_bf16_kernel<<<dim3(NS_ / 128, E_ / 128), 128, 0, stream>>>(
            ff1b, FF_, w2t, FF_, b2 + (size_t)l * E_, x1f, tmp, nullptr, E_, FF_, 0);

        layernorm_kernel<<<NS_, 256, 0, stream>>>(
            tmp, ln2g + (size_t)l * E_, ln2b + (size_t)l * E_, xf, xb);
    }

    cls_kernel<<<N_, 256, 0, stream>>>(xf, Wc, bc, (float*)d_out);
}
